// Joint_56487409877109
// MI455X (gfx1250) — compile-verified, matrix operand order checked by reading, not run on hardware
//
#include <hip/hip_runtime.h>

// ---------------------------------------------------------------------------
// Types for CDNA5 WMMA (gfx1250, wave32)
// ---------------------------------------------------------------------------
typedef __attribute__((ext_vector_type(16))) __bf16 bf16x16;
typedef __attribute__((ext_vector_type(8)))  float  f32x8;

struct U128x2 { uint4 a, b; };   // 32 bytes == bf16x16

static __device__ __forceinline__ unsigned short f2bf(float f) {
    unsigned u = __float_as_uint(f);
    unsigned r = (u + 0x7FFFu + ((u >> 16) & 1u)) >> 16;   // round-to-nearest-even
    return (unsigned short)r;
}

// CDNA5 async global->LDS copy (tracked by ASYNCcnt).  Each lane copies 16B:
//   LDS byte address in vdst, global address in 64-bit vaddr   (GV mode, SADDR=off)
static __device__ __forceinline__ void async_g2l_b128(const void* g, const void* l) {
    unsigned long long ga = (unsigned long long)g;
    unsigned la = (unsigned)(unsigned long long)l;   // low 32 bits = LDS byte offset
    asm volatile("global_load_async_to_lds_b128 %0, %1, off"
                 :: "v"(la), "v"(ga)
                 : "memory");
}
static __device__ __forceinline__ void wait_async0() {
    asm volatile("s_wait_asynccnt 0x0" ::: "memory");
}

// ---------------------------------------------------------------------------
// Tiled bf16 GEMM, f32 accumulate, V_WMMA_F32_16X16X32_BF16.
//   C[b] = act( A[b] (MxK row-major) @ Bt[b]^T + bias ),  Bt is N x K row-major.
// Block tile 128x128, 256 threads = 8 waves; wave w owns rows [w*16,w*16+16)
// and 8 n-tiles of 16.  K step 32.  Double-buffered LDS; BOTH operands staged
// with async global->LDS b128 copies (no ds_stores, no VALU packing in loop).
// Requires M%128==0, N%128==0, K%32==0.
// TBF16: write bf16 output transposed per tdim-sized batch: [m/tdim][N][tdim].
// ---------------------------------------------------------------------------
enum { ACT_NONE = 0, ACT_RELU = 1, ACT_CLIP = 2 };

template <int ACT, bool WF32, bool WBF16, bool TBF16>
__global__ __launch_bounds__(256)
void gemm_bf16_wmma(const unsigned short* __restrict__ A,
                    const unsigned short* __restrict__ Bt,
                    const float* __restrict__ bias,
                    float* __restrict__ Cf,
                    unsigned short* __restrict__ Cbf,
                    int N, int K, int tdim,
                    long long sA, long long sB, long long sC)
{
    constexpr int BM = 128, BN = 128, BK = 32, PAD = 8;
    constexpr int LDS_K = BK + PAD;                 // 40 elems -> 80B row stride (16B aligned)
    __shared__ alignas(16) unsigned short As[2][BM * LDS_K];   // 2 x 10240 B
    __shared__ alignas(16) unsigned short Bs[2][BN * LDS_K];   // 2 x 10240 B

    const int bz = blockIdx.z;
    A  += (long long)bz * sA;
    Bt += (long long)bz * sB;
    const long long cOff = (long long)bz * sC;

    const int m0   = blockIdx.y * BM;
    const int n0   = blockIdx.x * BN;
    const int tid  = threadIdx.x;
    const int lane = tid & 31;
    const int wave = tid >> 5;
    const int hi   = (lane >> 4) & 1;               // lane-group K split per ISA layout
    const int l16  = lane & 15;

    // staging coordinates: each thread copies 32B of A and 32B of B per k-tile
    const int srow = tid >> 1;                      // 0..127
    const int sseg = (tid & 1) * 16;                // 0 or 16

    f32x8 acc[8];
#pragma unroll
    for (int t = 0; t < 8; ++t)
#pragma unroll
        for (int e = 0; e < 8; ++e) acc[t][e] = 0.0f;

    const int kTiles = K / BK;

    auto issue = [&](int buf, int kt) {
        const unsigned short* srcA = A  + (long long)(m0 + srow) * K + kt * BK + sseg;
        const unsigned short* dstA = &As[buf][srow * LDS_K + sseg];
        async_g2l_b128(srcA,     dstA);
        async_g2l_b128(srcA + 8, dstA + 8);
        const unsigned short* srcB = Bt + (long long)(n0 + srow) * K + kt * BK + sseg;
        const unsigned short* dstB = &Bs[buf][srow * LDS_K + sseg];
        async_g2l_b128(srcB,     dstB);
        async_g2l_b128(srcB + 8, dstB + 8);
    };

    // ---- prologue: stage k-tile 0 into buffer 0 ----
    issue(0, 0);
    wait_async0();
    __syncthreads();

    for (int kt = 0; kt < kTiles; ++kt) {
        const int cur = kt & 1;
        const int nxt = cur ^ 1;
        const bool hasNext = (kt + 1) < kTiles;

        if (hasNext) issue(nxt, kt + 1);     // overlap next-tile DMA with WMMA

        // ---- A fragment: row l16, K chunks at hi*8 and 16+hi*8 ----
        U128x2 ar;
        {
            const unsigned short* base = &As[cur][(wave * 16 + l16) * LDS_K];
            ar.a = *(const uint4*)(base + hi * 8);
            ar.b = *(const uint4*)(base + 16 + hi * 8);
        }
        const bf16x16 av = __builtin_bit_cast(bf16x16, ar);

#pragma unroll
        for (int nt = 0; nt < 8; ++nt) {
            U128x2 br;
            const unsigned short* base = &Bs[cur][(nt * 16 + l16) * LDS_K];
            br.a = *(const uint4*)(base + hi * 8);
            br.b = *(const uint4*)(base + 16 + hi * 8);
            const bf16x16 bv = __builtin_bit_cast(bf16x16, br);
            acc[nt] = __builtin_amdgcn_wmma_f32_16x16x32_bf16(
                false, av, false, bv, (short)0, acc[nt], false, false);
        }

        if (hasNext) {
            wait_async0();                   // next tiles landed in LDS
            __syncthreads();                 // all waves done with `cur`, `nxt` visible
        }
    }

    // ---- epilogue: C layout VGPR r -> M = r + 8*hi, N = l16 ----
#pragma unroll
    for (int nt = 0; nt < 8; ++nt) {
        const int n  = n0 + nt * 16 + l16;
        const float bvs = bias ? bias[n] : 0.0f;
#pragma unroll
        for (int r = 0; r < 8; ++r) {
            const int m = m0 + wave * 16 + hi * 8 + r;
            float v = acc[nt][r] + bvs;
            if (ACT == ACT_RELU) v = v > 0.0f ? v : 0.0f;
            if (ACT == ACT_CLIP) v = fminf(fmaxf(v, -100.0f), 100.0f);
            const long long idx = cOff + (long long)m * N + n;
            if (WF32) Cf[idx] = v;
            if (WBF16) {
                if (TBF16) {
                    // transposed per-batch write: [m/tdim][N][tdim]
                    const long long ti =
                        ((long long)(m / tdim) * N + n) * tdim + (m % tdim);
                    Cbf[ti] = f2bf(v);
                } else {
                    Cbf[idx] = f2bf(v);
                }
            }
        }
    }
}

// ---------------------------------------------------------------------------
// LayerNorm over last dim (=1024), optional residual add; emits f32 and/or bf16.
// ---------------------------------------------------------------------------
__global__ __launch_bounds__(256)
void ln_kernel(const float* __restrict__ in, const float* __restrict__ res,
               const float* __restrict__ g, const float* __restrict__ beta,
               float* __restrict__ outf, unsigned short* __restrict__ outbf)
{
    constexpr int C = 1024;
    __shared__ float sm[256];
    const long long base = (long long)blockIdx.x * C;
    const int t = threadIdx.x;

    float v[4];
    float s = 0.0f;
#pragma unroll
    for (int i = 0; i < 4; ++i) {
        const int c = t + i * 256;
        float x = in[base + c];
        if (res) x += res[base + c];
        v[i] = x; s += x;
    }
    sm[t] = s; __syncthreads();
    for (int o = 128; o > 0; o >>= 1) { if (t < o) sm[t] += sm[t + o]; __syncthreads(); }
    const float mean = sm[0] * (1.0f / C); __syncthreads();

    float s2 = 0.0f;
#pragma unroll
    for (int i = 0; i < 4; ++i) { const float d = v[i] - mean; s2 += d * d; }
    sm[t] = s2; __syncthreads();
    for (int o = 128; o > 0; o >>= 1) { if (t < o) sm[t] += sm[t + o]; __syncthreads(); }
    const float rstd = rsqrtf(sm[0] * (1.0f / C) + 1e-5f);

#pragma unroll
    for (int i = 0; i < 4; ++i) {
        const int c = t + i * 256;
        const float y = (v[i] - mean) * rstd * g[c] + beta[c];
        if (outf)  outf[base + c]  = y;
        if (outbf) outbf[base + c] = f2bf(y);
    }
}

// ---------------------------------------------------------------------------
// Masked, scaled softmax over last dim (=1024); writes bf16 probabilities.
// ---------------------------------------------------------------------------
__global__ __launch_bounds__(256)
void softmax_mask_kernel(const float* __restrict__ scores, const int* __restrict__ mask,
                         const float* __restrict__ scale_p,
                         unsigned short* __restrict__ attn, int S)
{
    __shared__ float sm[256];
    const int row = blockIdx.x;             // 0 .. B*S-1
    const int b   = row / S;
    const int t   = threadIdx.x;
    const float scale = scale_p[0];
    const float* src = scores + (long long)row * S;
    const int*  mrow = mask + (long long)b * S;

    float v[4];
    float mx = -3.0e38f;
#pragma unroll
    for (int i = 0; i < 4; ++i) {
        const int c = t + i * 256;
        float x = src[c] * scale;
        if (mrow[c] == 0) x = -1.0e9f;
        v[i] = x; mx = fmaxf(mx, x);
    }
    sm[t] = mx; __syncthreads();
    for (int o = 128; o > 0; o >>= 1) { if (t < o) sm[t] = fmaxf(sm[t], sm[t + o]); __syncthreads(); }
    mx = sm[0]; __syncthreads();

    float s = 0.0f;
#pragma unroll
    for (int i = 0; i < 4; ++i) { v[i] = __expf(v[i] - mx); s += v[i]; }
    sm[t] = s; __syncthreads();
    for (int o = 128; o > 0; o >>= 1) { if (t < o) sm[t] += sm[t + o]; __syncthreads(); }
    const float inv = 1.0f / sm[0];

    unsigned short* dst = attn + (long long)row * S;
#pragma unroll
    for (int i = 0; i < 4; ++i) { const int c = t + i * 256; dst[c] = f2bf(v[i] * inv); }
}

// ---------------------------------------------------------------------------
// fp32 [K][N]  ->  bf16 [N][K]  tiled transpose+convert (32x32 LDS tiles).
// Grid: (N/32, K/32), 256 threads (32x8).
// ---------------------------------------------------------------------------
__global__ __launch_bounds__(256)
void cvt_transpose_f32_bf16(const float* __restrict__ in, unsigned short* __restrict__ out,
                            int K, int N)
{
    __shared__ unsigned short tile[32][33];
    const int n0 = blockIdx.x * 32;
    const int k0 = blockIdx.y * 32;
    const int tx = threadIdx.x & 31;
    const int ty = threadIdx.x >> 5;     // 0..7
#pragma unroll
    for (int i = 0; i < 32; i += 8)
        tile[ty + i][tx] = f2bf(in[(long long)(k0 + ty + i) * N + n0 + tx]);
    __syncthreads();
#pragma unroll
    for (int i = 0; i < 32; i += 8)
        out[(long long)(n0 + ty + i) * K + k0 + tx] = tile[tx][ty + i];
}

// ---------------------------------------------------------------------------
// Host-side orchestration
// ---------------------------------------------------------------------------
extern "C" void kernel_launch(void* const* d_in, const int* in_sizes, int n_in,
                              void* d_out, int out_size, void* d_ws, size_t ws_size,
                              hipStream_t stream)
{
    (void)in_sizes; (void)n_in; (void)out_size; (void)ws_size;

    const int Bn = 16, S = 1024, Din = 1024, Dh = 4096, Dout = 1024;
    const long long R = (long long)Bn * S;     // 16384 rows

    // inputs (setup_inputs dict order)
    const float* x        = (const float*)d_in[0];
    const int*   mask     = (const int*)  d_in[1];
    const float* ln_in_g  = (const float*)d_in[2];
    const float* ln_in_b  = (const float*)d_in[3];
    const float* W_mlp    = (const float*)d_in[4];
    const float* b_mlp    = (const float*)d_in[5];
    const float* W_proj   = (const float*)d_in[6];
    const float* b_proj   = (const float*)d_in[7];
    const float* scale    = (const float*)d_in[8];
    const float* Wq       = (const float*)d_in[9];
    const float* bq       = (const float*)d_in[10];
    const float* Wk       = (const float*)d_in[11];
    const float* bk       = (const float*)d_in[12];
    const float* Wv       = (const float*)d_in[13];
    const float* bv       = (const float*)d_in[14];
    const float* ln1_g    = (const float*)d_in[15];
    const float* ln1_b    = (const float*)d_in[16];
    const float* W_f1     = (const float*)d_in[17];
    const float* b_f1     = (const float*)d_in[18];
    const float* W_f2     = (const float*)d_in[19];
    const float* b_f2     = (const float*)d_in[20];
    const float* ln2_g    = (const float*)d_in[21];
    const float* ln2_b    = (const float*)d_in[22];
    const float* ln_out_g = (const float*)d_in[23];
    const float* ln_out_b = (const float*)d_in[24];

    // workspace carve-out
    char* ws = (char*)d_ws;
    size_t off = 0;
    auto alloc = [&](size_t bytes) -> void* {
        void* p = ws + off;
        off += (bytes + 255) & ~(size_t)255;
        return p;
    };
    // all weight copies stored TRANSPOSED: [N][K] bf16
    unsigned short* xln_bf = (unsigned short*)alloc((size_t)R * Din * 2);
    unsigned short* wMLPt  = (unsigned short*)alloc((size_t)Din * Dh * 2);     // [Dh][Din]
    unsigned short* wPROJt = (unsigned short*)alloc((size_t)Dh * Dout * 2);    // [Dout][Dh]
    unsigned short* wQt    = (unsigned short*)alloc((size_t)Dout * Dout * 2);
    unsigned short* wKt    = (unsigned short*)alloc((size_t)Dout * Dout * 2);
    unsigned short* wVt    = (unsigned short*)alloc((size_t)Dout * Dout * 2);
    unsigned short* wF1t   = (unsigned short*)alloc((size_t)Dout * 4 * Dout * 2); // [4D][D]
    unsigned short* wF2t   = (unsigned short*)alloc((size_t)4 * Dout * Dout * 2); // [D][4D]
    unsigned short* h_bf   = (unsigned short*)alloc((size_t)R * Dh * 2);   // reused for ffn1
    float*          xp_f   = (float*)         alloc((size_t)R * Dout * 4);
    unsigned short* xp_bf  = (unsigned short*)alloc((size_t)R * Dout * 2);
    unsigned short* q_bf   = (unsigned short*)alloc((size_t)R * Dout * 2);
    unsigned short* k_bf   = (unsigned short*)alloc((size_t)R * Dout * 2);
    unsigned short* vT_bf  = (unsigned short*)alloc((size_t)R * Dout * 2);  // [B][Dout][S]
    float*          sc_f   = (float*)         alloc((size_t)Bn * S * S * 4); // scores -> attn_out -> x2
    unsigned short* at_bf  = (unsigned short*)alloc((size_t)Bn * S * S * 2);
    float*          x1_f   = (float*)         alloc((size_t)R * Dout * 4);
    unsigned short* x1_bf  = (unsigned short*)alloc((size_t)R * Dout * 2);
    float*          ffn2_f = (float*)         alloc((size_t)R * Dout * 4);

    // weight conversions fp32 [K][N] -> bf16 [N][K]
    cvt_transpose_f32_bf16<<<dim3(Dh / 32, Din / 32), 256, 0, stream>>>(W_mlp,  wMLPt,  Din, Dh);
    cvt_transpose_f32_bf16<<<dim3(Dout / 32, Dh / 32), 256, 0, stream>>>(W_proj, wPROJt, Dh, Dout);
    cvt_transpose_f32_bf16<<<dim3(Dout / 32, Dout / 32), 256, 0, stream>>>(Wq, wQt, Dout, Dout);
    cvt_transpose_f32_bf16<<<dim3(Dout / 32, Dout / 32), 256, 0, stream>>>(Wk, wKt, Dout, Dout);
    cvt_transpose_f32_bf16<<<dim3(Dout / 32, Dout / 32), 256, 0, stream>>>(Wv, wVt, Dout, Dout);
    cvt_transpose_f32_bf16<<<dim3(4 * Dout / 32, Dout / 32), 256, 0, stream>>>(W_f1, wF1t, Dout, 4 * Dout);
    cvt_transpose_f32_bf16<<<dim3(Dout / 32, 4 * Dout / 32), 256, 0, stream>>>(W_f2, wF2t, 4 * Dout, Dout);

    // 1) input LN -> bf16
    ln_kernel<<<(unsigned)R, 256, 0, stream>>>(x, nullptr, ln_in_g, ln_in_b, nullptr, xln_bf);

    // 2) h = relu(xln @ W_mlp + b_mlp)   [R,4096] bf16
    gemm_bf16_wmma<ACT_RELU, false, true, false>
        <<<dim3(Dh / 128, R / 128, 1), 256, 0, stream>>>(
            xln_bf, wMLPt, b_mlp, nullptr, h_bf, Dh, Din, 1, 0, 0, 0);

    // 3) xp = clip(h @ W_proj + b_proj)  [R,1024] f32 + bf16
    gemm_bf16_wmma<ACT_CLIP, true, true, false>
        <<<dim3(Dout / 128, R / 128, 1), 256, 0, stream>>>(
            h_bf, wPROJt, b_proj, xp_f, xp_bf, Dout, Dh, 1, 0, 0, 0);

    // 4) q, k (row-major bf16) and v (transposed per-batch [B][Dout][S])
    gemm_bf16_wmma<ACT_NONE, false, true, false>
        <<<dim3(Dout / 128, R / 128, 1), 256, 0, stream>>>(
            xp_bf, wQt, bq, nullptr, q_bf, Dout, Dout, 1, 0, 0, 0);
    gemm_bf16_wmma<ACT_NONE, false, true, false>
        <<<dim3(Dout / 128, R / 128, 1), 256, 0, stream>>>(
            xp_bf, wKt, bk, nullptr, k_bf, Dout, Dout, 1, 0, 0, 0);
    gemm_bf16_wmma<ACT_NONE, false, true, true>
        <<<dim3(Dout / 128, R / 128, 1), 256, 0, stream>>>(
            xp_bf, wVt, bv, nullptr, vT_bf, Dout, Dout, S, 0, 0, 0);

    // 5) scores = q @ k^T  (batched; k [S,Dout] row-major is already [N][K])
    gemm_bf16_wmma<ACT_NONE, true, false, false>
        <<<dim3(S / 128, S / 128, Bn), 256, 0, stream>>>(
            q_bf, k_bf, nullptr, sc_f, nullptr, S, Dout, 1,
            (long long)S * Dout, (long long)S * Dout, (long long)S * S);

    // 6) attn = softmax(scale * scores + mask) -> bf16
    softmax_mask_kernel<<<(unsigned)(Bn * S), 256, 0, stream>>>(sc_f, mask, scale, at_bf, S);

    // 7) attn_out = attn @ v   (batched; vT [Dout][S] per batch is [N][K])
    float* attn_out = sc_f;
    gemm_bf16_wmma<ACT_NONE, true, false, false>
        <<<dim3(Dout / 128, S / 128, Bn), 256, 0, stream>>>(
            at_bf, vT_bf, nullptr, attn_out, nullptr, Dout, S, 1,
            (long long)S * S, (long long)Dout * S, (long long)S * Dout);

    // 8) x1 = LN(xp + attn_out)
    ln_kernel<<<(unsigned)R, 256, 0, stream>>>(xp_f, attn_out, ln1_g, ln1_b, x1_f, x1_bf);

    // 9) ffn1 = relu(x1 @ W_f1 + b_f1)  [R,4096] bf16 (reuse h_bf)
    gemm_bf16_wmma<ACT_RELU, false, true, false>
        <<<dim3(4 * Dout / 128, R / 128, 1), 256, 0, stream>>>(
            x1_bf, wF1t, b_f1, nullptr, h_bf, 4 * Dout, Dout, 1, 0, 0, 0);

    // 10) ffn2 = ffn1 @ W_f2 + b_f2  [R,1024] f32
    gemm_bf16_wmma<ACT_NONE, true, false, false>
        <<<dim3(Dout / 128, R / 128, 1), 256, 0, stream>>>(
            h_bf, wF2t, b_f2, ffn2_f, nullptr, Dout, 4 * Dout, 1, 0, 0, 0);

    // 11) x2 = LN(x1 + ffn2)  (reuse scores buffer for x2)
    float* x2_f = sc_f;
    ln_kernel<<<(unsigned)R, 256, 0, stream>>>(x1_f, ffn2_f, ln2_g, ln2_b, x2_f, nullptr);

    // 12) out = LN(x2)
    ln_kernel<<<(unsigned)R, 256, 0, stream>>>(x2_f, nullptr, ln_out_g, ln_out_b,
                                               (float*)d_out, nullptr);
}
